// Edge_node_50869592655555
// MI455X (gfx1250) — matrix operand order repaired and checked by reading, not compiled, not run on hardware
//
#include <hip/hip_runtime.h>

typedef __attribute__((ext_vector_type(16))) __bf16 v16bf;
typedef __attribute__((ext_vector_type(8)))  __bf16 v8bf;
typedef __attribute__((ext_vector_type(8)))  float  v8f;

#define WVN 4     // waves per block, node kernel
#define HS 264    // hidden LDS row stride in bf16 (132 dwords -> 4l+j bank pattern, conflict-free)
#define SBS 256   // scan block size

static __device__ __forceinline__ v8f wmma_bf16(v16bf a, v16bf b, v8f c) {
    // D = A(16x32 bf16) * B(32x16 bf16) + C(16x16 f32)
    return __builtin_amdgcn_wmma_f32_16x16x32_bf16(false, a, false, b, (short)0, c, false, false);
}

static __device__ __forceinline__ v8f splat8(float x) {
    v8f v = {x, x, x, x, x, x, x, x};
    return v;
}

// Convert one 8-float global chunk to 8 bf16 elements of an A fragment half.
static __device__ __forceinline__ void cvt_chunk(const float* __restrict__ p, __bf16* e) {
    float4 x = *(const float4*)p;
    float4 y = *(const float4*)(p + 4);
    e[0] = (__bf16)x.x; e[1] = (__bf16)x.y; e[2] = (__bf16)x.z; e[3] = (__bf16)x.w;
    e[4] = (__bf16)y.x; e[5] = (__bf16)y.y; e[6] = (__bf16)y.z; e[7] = (__bf16)y.w;
}

// Pack f32 weight [K x Nout] into bf16 B-fragment layout:
// out[(kt*Nout + n)*32 + 16*((kk>>3)&1) + 8*(kk>>4) + (kk&7)] = W[kt*32 + kk][n]
__global__ void pack_w(const float* __restrict__ W, __bf16* __restrict__ out,
                       int K, int Nout) {
    int tid = blockIdx.x * blockDim.x + threadIdx.x;
    if (tid >= K * Nout) return;
    int k = tid / Nout, n = tid - k * Nout;
    int kt = k >> 5, kk = k & 31;
    out[(((size_t)(kt * Nout + n)) << 5) + (((kk >> 3) & 1) << 4) + ((kk >> 4) << 3) + (kk & 7)] = (__bf16)W[(size_t)k * Nout + n];
}

// ---------------- CSR build: count -> exclusive scan -> fill ----------------
__global__ void count_kernel(const int* __restrict__ esrc, const int* __restrict__ edst,
                             int* __restrict__ count, int E) {
    int i = blockIdx.x * blockDim.x + threadIdx.x;
    if (i >= 2 * E) return;
    int node = (i < E) ? esrc[i] : edst[i - E];
    atomicAdd(&count[node], 1);
}

__global__ void scan1_kernel(const int* __restrict__ count, int* __restrict__ offsets,
                             int* __restrict__ bsum, int n) {
    __shared__ int sh[SBS];
    int i = blockIdx.x * SBS + threadIdx.x;
    int c = (i < n) ? count[i] : 0;
    sh[threadIdx.x] = c;
    __syncthreads();
    for (int off = 1; off < SBS; off <<= 1) {             // Hillis-Steele inclusive scan
        int t = (threadIdx.x >= off) ? sh[threadIdx.x - off] : 0;
        __syncthreads();
        sh[threadIdx.x] += t;
        __syncthreads();
    }
    if (i < n) offsets[i] = sh[threadIdx.x] - c;          // exclusive within block
    if (threadIdx.x == SBS - 1) bsum[blockIdx.x] = sh[SBS - 1];
}

__global__ void scan2_kernel(int* __restrict__ bsum, int nb) {  // single block of 512
    __shared__ int sh[512];
    int c = (threadIdx.x < nb) ? bsum[threadIdx.x] : 0;
    sh[threadIdx.x] = c;
    __syncthreads();
    for (int off = 1; off < 512; off <<= 1) {
        int t = (threadIdx.x >= off) ? sh[threadIdx.x - off] : 0;
        __syncthreads();
        sh[threadIdx.x] += t;
        __syncthreads();
    }
    if (threadIdx.x < nb) bsum[threadIdx.x] = sh[threadIdx.x] - c;  // exclusive
}

__global__ void scan3_kernel(int* __restrict__ offsets, const int* __restrict__ bsum, int n) {
    int i = blockIdx.x * SBS + threadIdx.x;
    if (i < n) offsets[i] += bsum[blockIdx.x];
}

// Fill uses offsets as the running cursor; afterwards offsets[node] == end-of-list.
__global__ void fill_kernel(const int* __restrict__ esrc, const int* __restrict__ edst,
                            int* __restrict__ offsets, int* __restrict__ entries, int E) {
    int i = blockIdx.x * blockDim.x + threadIdx.x;
    if (i >= 2 * E) return;
    int node = (i < E) ? esrc[i] : edst[i - E];
    int pos = atomicAdd(&offsets[node], 1);
    entries[pos] = (i < E) ? i : (i - E);
}

// ---------------- edge MLP, M=32 edges/wave: [attr | x[src] | x[dst]] (384) -> 256 relu -> 128 ----------------
__global__ __launch_bounds__(64)
void edge_kernel(const float* __restrict__ node_rep, const float* __restrict__ edge_attr,
                 const int* __restrict__ esrc, const int* __restrict__ edst,
                 const float* __restrict__ b1, const float* __restrict__ b2,
                 const __bf16* __restrict__ w1p, const __bf16* __restrict__ w2p,
                 float* __restrict__ edge_out, int ntiles) {
    __shared__ __bf16 hid[2][32 * HS];

    int wave = threadIdx.x >> 5;
    int lane = threadIdx.x & 31;
    int tile = blockIdx.x * 2 + wave;
    if (tile >= ntiles) return;
    int r = lane & 15, half = lane >> 4;
    int e0 = tile << 5;

    // Gather A fragments for both 16-row sub-tiles, all 12 k-tiles (K=384).
    v16bf A0[12], A1[12];
#pragma unroll
    for (int m = 0; m < 2; ++m) {
        int erow = e0 + m * 16 + r;
        int s = esrc[erow], d = edst[erow];
        const float* arow = edge_attr + (size_t)erow * 128;
        const float* srow = node_rep + (size_t)s * 128;
        const float* drow = node_rep + (size_t)d * 128;
#pragma unroll
        for (int kt = 0; kt < 12; ++kt) {
            union { v16bf v; __bf16 e[16]; } u;
#pragma unroll
            for (int c = 0; c < 2; ++c) {
                int kb = kt * 32 + 8 * half + 16 * c;
                const float* p = (kb < 128) ? (arow + kb)
                               : (kb < 256) ? (srow + (kb - 128))
                                            : (drow + (kb - 256));
                cvt_chunk(p, &u.e[8 * c]);
            }
            if (m) A1[kt] = u.v; else A0[kt] = u.v;
        }
    }

    // Layer 1: each B fragment feeds two WMMA chains (2x arithmetic intensity on L2 weights).
    __bf16* h = hid[wave];
    for (int jn = 0; jn < 16; ++jn) {
        int n0 = jn * 16;
        float bv = b1[n0 + r];
        v8f acc0 = splat8(bv), acc1 = splat8(bv);
#pragma unroll
        for (int kt = 0; kt < 12; ++kt) {
            v16bf B = *(const v16bf*)(w1p + (((size_t)(kt * 256 + n0 + r)) << 5) + half * 16);
            acc0 = wmma_bf16(A0[kt], B, acc0);
            acc1 = wmma_bf16(A1[kt], B, acc1);
        }
#pragma unroll
        for (int v = 0; v < 8; ++v) {
            int row = v + 8 * half;
            float x0 = acc0[v]; x0 = x0 > 0.f ? x0 : 0.f;
            float x1 = acc1[v]; x1 = x1 > 0.f ? x1 : 0.f;
            h[row * HS + n0 + r] = (__bf16)x0;          // C-layout -> row-major LDS
            h[(16 + row) * HS + n0 + r] = (__bf16)x1;
        }
    }

    // Re-load hidden in A layout (per-wave DS ordering guarantees store->load).
    v16bf A2a[8], A2b[8];
#pragma unroll
    for (int kt = 0; kt < 8; ++kt) {
        const __bf16* p0 = h + r * HS + kt * 32 + 8 * half;
        const __bf16* p1 = h + (16 + r) * HS + kt * 32 + 8 * half;
        union { v16bf v; struct { v8bf lo; v8bf hi; } s; } u0, u1;
        u0.s.lo = *(const v8bf*)p0; u0.s.hi = *(const v8bf*)(p0 + 16);
        u1.s.lo = *(const v8bf*)p1; u1.s.hi = *(const v8bf*)(p1 + 16);
        A2a[kt] = u0.v; A2b[kt] = u1.v;
    }

    // Layer 2: out[32x128] = hidden @ W2 + b2 (no scatter here any more).
    for (int jo = 0; jo < 8; ++jo) {
        int n0 = jo * 16;
        float bv = b2[n0 + r];
        v8f acc0 = splat8(bv), acc1 = splat8(bv);
#pragma unroll
        for (int kt = 0; kt < 8; ++kt) {
            v16bf B = *(const v16bf*)(w2p + (((size_t)(kt * 128 + n0 + r)) << 5) + half * 16);
            acc0 = wmma_bf16(A2a[kt], B, acc0);
            acc1 = wmma_bf16(A2b[kt], B, acc1);
        }
#pragma unroll
        for (int v = 0; v < 8; ++v) {
            int row = v + 8 * half;
            edge_out[(size_t)(e0 + row) * 128 + n0 + r] = acc0[v];
            edge_out[(size_t)(e0 + 16 + row) * 128 + n0 + r] = acc1[v];
        }
    }
}

// ---------------- gather-sum: edge2node[node] = sum of edge_out over incident edges ----------------
__global__ __launch_bounds__(256)
void gather_kernel(const float* __restrict__ edge_out, const int* __restrict__ entries,
                   const int* __restrict__ offsets /*end*/, const int* __restrict__ count,
                   float* __restrict__ nodebuf, int N) {
    int wave = threadIdx.x >> 5;
    int lane = threadIdx.x & 31;
    int node = blockIdx.x * (blockDim.x >> 5) + wave;
    if (node >= N) return;
    int end = offsets[node];
    int start = end - count[node];
    float a0 = 0.f, a1 = 0.f, a2 = 0.f, a3 = 0.f;
    for (int j = start; j < end; ++j) {
        const float* p = edge_out + (size_t)entries[j] * 128 + lane;
        a0 += p[0]; a1 += p[32]; a2 += p[64]; a3 += p[96];
    }
    float* o = nodebuf + (size_t)node * 128 + lane;
    o[0] = a0; o[32] = a1; o[64] = a2; o[96] = a3;
}

// ---------------- node MLP: [node_rep | edge2node] (256) -> 256 relu -> 128 (in-place on nodebuf) ----------------
__global__ __launch_bounds__(32 * WVN)
void node_kernel(const float* __restrict__ node_rep,
                 const float* __restrict__ b1, const float* __restrict__ b2,
                 const __bf16* __restrict__ w1p, const __bf16* __restrict__ w2p,
                 float* __restrict__ nodebuf, int ntiles) {
    __shared__ __bf16 hid[WVN][16 * HS];

    int wave = threadIdx.x >> 5;
    int lane = threadIdx.x & 31;
    int tile = blockIdx.x * WVN + wave;
    if (tile >= ntiles) return;
    int r = lane & 15, half = lane >> 4;
    int n0i = tile << 4;
    int node = n0i + r;

    const float* xr = node_rep + (size_t)node * 128;
    const float* er = nodebuf + (size_t)node * 128;   // gathered edge2node (read before overwrite)

    v16bf A[8];
#pragma unroll
    for (int kt = 0; kt < 8; ++kt) {
        union { v16bf v; __bf16 e[16]; } u;
#pragma unroll
        for (int c = 0; c < 2; ++c) {
            int kb = kt * 32 + 8 * half + 16 * c;
            const float* p = (kb < 128) ? (xr + kb) : (er + (kb - 128));
            cvt_chunk(p, &u.e[8 * c]);
        }
        A[kt] = u.v;
    }

    __bf16* h = hid[wave];
    for (int jn = 0; jn < 16; ++jn) {
        int n0 = jn * 16;
        v8f acc = splat8(b1[n0 + r]);
#pragma unroll
        for (int kt = 0; kt < 8; ++kt) {
            v16bf B = *(const v16bf*)(w1p + (((size_t)(kt * 256 + n0 + r)) << 5) + half * 16);
            acc = wmma_bf16(A[kt], B, acc);
        }
#pragma unroll
        for (int v = 0; v < 8; ++v) {
            float x = acc[v];
            x = x > 0.f ? x : 0.f;
            h[(v + 8 * half) * HS + n0 + r] = (__bf16)x;
        }
    }

    v16bf A2[8];
#pragma unroll
    for (int kt = 0; kt < 8; ++kt) {
        const __bf16* p = h + r * HS + kt * 32 + 8 * half;
        union { v16bf v; struct { v8bf lo; v8bf hi; } s; } u;
        u.s.lo = *(const v8bf*)p;
        u.s.hi = *(const v8bf*)(p + 16);
        A2[kt] = u.v;
    }

    for (int jo = 0; jo < 8; ++jo) {
        int n0 = jo * 16;
        v8f acc = splat8(b2[n0 + r]);
#pragma unroll
        for (int kt = 0; kt < 8; ++kt) {
            v16bf B = *(const v16bf*)(w2p + (((size_t)(kt * 128 + n0 + r)) << 5) + half * 16);
            acc = wmma_bf16(A2[kt], B, acc);
        }
#pragma unroll
        for (int v = 0; v < 8; ++v) {
            int row = v + 8 * half;
            nodebuf[(size_t)(n0i + row) * 128 + n0 + r] = acc[v];
        }
    }
}

extern "C" void kernel_launch(void* const* d_in, const int* in_sizes, int n_in,
                              void* d_out, int out_size, void* d_ws, size_t ws_size,
                              hipStream_t stream) {
    const float* node_rep  = (const float*)d_in[0];
    const float* edge_attr = (const float*)d_in[1];
    const int*   esrc      = (const int*)d_in[2];
    const int*   edst      = (const int*)d_in[3];
    const float* W1e = (const float*)d_in[4];
    const float* b1e = (const float*)d_in[5];
    const float* W2e = (const float*)d_in[6];
    const float* b2e = (const float*)d_in[7];
    const float* W1n = (const float*)d_in[8];
    const float* b1n = (const float*)d_in[9];
    const float* W2n = (const float*)d_in[10];
    const float* b2n = (const float*)d_in[11];

    const int R = 128;
    int N = in_sizes[0] / R;
    int E = in_sizes[2];

    float* node_out = (float*)d_out;            // holds edge2node, then node_out in place
    float* edge_out = node_out + (size_t)N * R;

    // Workspace layout
    char* ws = (char*)d_ws;
    __bf16* w1e_p = (__bf16*)ws;                        // 384*256 bf16
    __bf16* w2e_p = w1e_p + 98304;                      // 256*128
    __bf16* w1n_p = w1e_p + 131072;                     // 256*256
    __bf16* w2n_p = w1e_p + 196608;                     // 256*128 -> 458752 B total
    int* count   = (int*)(ws + 458752);                 // N ints
    int* offsets = count + N;                           // N ints
    int* bsum    = offsets + N;                         // 512 ints
    int* entries = bsum + 512;                          // 2E ints

    // Weight packing (bf16 B-fragment layout)
    const int T = 256;
    pack_w<<<(384 * 256 + T - 1) / T, T, 0, stream>>>(W1e, w1e_p, 384, 256);
    pack_w<<<(256 * 128 + T - 1) / T, T, 0, stream>>>(W2e, w2e_p, 256, 128);
    pack_w<<<(256 * 256 + T - 1) / T, T, 0, stream>>>(W1n, w1n_p, 256, 256);
    pack_w<<<(256 * 128 + T - 1) / T, T, 0, stream>>>(W2n, w2n_p, 256, 128);

    // CSR build for the scatter -> gather conversion
    hipMemsetAsync(count, 0, (size_t)N * sizeof(int), stream);
    count_kernel<<<(2 * E + T - 1) / T, T, 0, stream>>>(esrc, edst, count, E);
    int nb = (N + SBS - 1) / SBS;   // 391 for N=100000, fits single 512-thread scan block
    scan1_kernel<<<nb, SBS, 0, stream>>>(count, offsets, bsum, N);
    scan2_kernel<<<1, 512, 0, stream>>>(bsum, nb);
    scan3_kernel<<<nb, SBS, 0, stream>>>(offsets, bsum, N);
    fill_kernel<<<(2 * E + T - 1) / T, T, 0, stream>>>(esrc, edst, offsets, entries, E);

    // Edge MLP (WMMA), then gather-sum, then node MLP (WMMA)
    int etiles = (E + 31) / 32;
    edge_kernel<<<(etiles + 1) / 2, 64, 0, stream>>>(
        node_rep, edge_attr, esrc, edst, b1e, b2e, w1e_p, w2e_p, edge_out, etiles);
    gather_kernel<<<(N + 7) / 8, 256, 0, stream>>>(edge_out, entries, offsets, count, node_out, N);
    int ntiles = (N + 15) / 16;
    node_kernel<<<(ntiles + WVN - 1) / WVN, 32 * WVN, 0, stream>>>(
        node_rep, b1n, b2n, w1n_p, w2n_p, node_out, ntiles);
}
